// TransformerBlock_91139206021482
// MI455X (gfx1250) — matrix operand order repaired and checked by reading, not compiled, not run on hardware
//
#include <hip/hip_runtime.h>
#include <hip/hip_bf16.h>
#include <math.h>

// ---------------------------------------------------------------------------
// Types for CDNA5 WMMA (wave32): v_wmma_f32_16x16x32_bf16
// ---------------------------------------------------------------------------
typedef __bf16 bf16_t;
typedef __attribute__((ext_vector_type(16))) __bf16 v16bf;
typedef __attribute__((ext_vector_type(8)))  __bf16 v8bf;
typedef __attribute__((ext_vector_type(8)))  float  v8f;

#define BB    4
#define TT    4096
#define CCH   768
#define DFFH  3072
#define C3H   2304      // 3*C
#define BTH   16384     // B*T
#define ATT_SCALE 0.03608439182435161f  // 1/sqrt(768)

// D = A(16x32 bf16) * B(32x16 bf16) + C(16x16 f32)
__device__ __forceinline__ v8f wmma_bf16(v16bf a, v16bf b, v8f c) {
  return __builtin_amdgcn_wmma_f32_16x16x32_bf16(false, a, false, b, (short)0, c,
                                                 false, false);
}

// ---------------------------------------------------------------------------
// Fragment loaders, contiguous k order used here (k = 16*(lane>=16) + s), same for A and B
// A 16x32 bf16: lane M = lane&15 ; VGPR0-3 K = kb+0..7, VGPR4-7 K = kb+8..15,
//               kb = 16*(lane>=16)  -> two 16B chunks per lane
// B 32x16 bf16 from W[N,K] row-major: lane N = lane&15 ; VGPR v K = kh+2v,2v+1,
//               kh = 16*(lane>=16) -> one 32B chunk per lane
// C/D f32 16x16: row = vgpr + 8*(lane>=16), col = lane&15
// ---------------------------------------------------------------------------
__device__ __forceinline__ v16bf load_a_frag(const bf16_t* __restrict__ A, int lda,
                                             int row0, int k0, int lane) {
  const int m  = row0 + (lane & 15);
  const int kb = (lane >> 4) << 4;
  const bf16_t* p = A + (size_t)m * lda + (size_t)(k0 + kb);
  v8bf c0 = *(const v8bf*)(p);
  v8bf c1 = *(const v8bf*)(p + 8);
  v16bf r;
#pragma unroll
  for (int i = 0; i < 8; ++i) { r[i] = c0[i]; r[8 + i] = c1[i]; }
  return r;
}

__device__ __forceinline__ v16bf load_b_frag(const bf16_t* __restrict__ W, int ldw,
                                             int col0, int k0, int lane) {
  const int n  = col0 + (lane & 15);
  const int kh = (lane >> 4) << 4;
  return *(const v16bf*)(W + (size_t)n * ldw + (size_t)(k0 + kh));
}

// Same layouts, sourced from an LDS tile stored [128 rows][32 k] bf16
__device__ __forceinline__ v16bf lds_a_frag(const bf16_t* s, int row0, int lane) {
  const int m  = row0 + (lane & 15);
  const int kb = (lane >> 4) << 4;
  const bf16_t* p = s + m * 32 + kb;
  v8bf c0 = *(const v8bf*)(p);
  v8bf c1 = *(const v8bf*)(p + 8);
  v16bf r;
#pragma unroll
  for (int i = 0; i < 8; ++i) { r[i] = c0[i]; r[8 + i] = c1[i]; }
  return r;
}

__device__ __forceinline__ v16bf lds_b_frag(const bf16_t* s, int n0, int lane) {
  const int n  = n0 + (lane & 15);
  const int kh = (lane >> 4) << 4;
  return *(const v16bf*)(s + n * 32 + kh);
}

// ---------------------------------------------------------------------------
// Async global -> LDS staging of a 128x32 bf16 tile (8KB): 512 x 16B chunks,
// 2 per thread with 256 threads.  Uses the CDNA5 async-copy path (ASYNCcnt).
// LDS byte offset = low 32 bits of the generic shared-memory address
// (LDS aperture maps addr[31:0] to the LDS offset).
// ---------------------------------------------------------------------------
__device__ __forceinline__ void stage_tile_async(const bf16_t* __restrict__ gbase,
                                                 int ld, int rowBase, int k,
                                                 bf16_t* sdst, int tid) {
#pragma unroll
  for (int h = 0; h < 2; ++h) {
    const int id  = tid + h * 256;
    const int row = id >> 2;
    const int cc  = (id & 3) * 8;               // element offset of 16B chunk
    const bf16_t* g = gbase + (size_t)(rowBase + row) * ld + (size_t)(k + cc);
    const unsigned lofs = (unsigned)(size_t)(sdst + row * 32 + cc);
    asm volatile("global_load_async_to_lds_b128 %0, %1, off"
                 :: "v"(lofs), "v"(g) : "memory");
  }
}

__device__ __forceinline__ void wait_async_all() {
  asm volatile("s_wait_asynccnt 0" ::: "memory");
}

// 16-lane group reductions (cols of a C/D tile live in one lane half)
__device__ __forceinline__ float redmax16(float v) {
#pragma unroll
  for (int off = 1; off < 16; off <<= 1) v = fmaxf(v, __shfl_xor(v, off, 32));
  return v;
}
__device__ __forceinline__ float redsum16(float v) {
#pragma unroll
  for (int off = 1; off < 16; off <<= 1) v += __shfl_xor(v, off, 32);
  return v;
}

// ---------------------------------------------------------------------------
// f32 -> bf16 conversion
// ---------------------------------------------------------------------------
__global__ __launch_bounds__(256) void cvt_f32_bf16(const float* __restrict__ src,
                                                    bf16_t* __restrict__ dst, int n) {
  int i = blockIdx.x * 256 + threadIdx.x;
  if (i < n) dst[i] = (bf16_t)src[i];
}

// ---------------------------------------------------------------------------
// LayerNorm: one block (256 thr) per row of C=768, write bf16
// ---------------------------------------------------------------------------
__global__ __launch_bounds__(256) void layernorm_bf16(const float* __restrict__ x,
                                                      const float* __restrict__ g,
                                                      const float* __restrict__ b,
                                                      bf16_t* __restrict__ out) {
  __shared__ float s1[256], s2[256];
  const int row = blockIdx.x;
  const int tid = threadIdx.x;
  const float* xr = x + (size_t)row * CCH;
  float a0 = 0.f, a1 = 0.f;
  for (int c = tid; c < CCH; c += 256) { float v = xr[c]; a0 += v; a1 += v * v; }
  s1[tid] = a0; s2[tid] = a1;
  __syncthreads();
  for (int off = 128; off > 0; off >>= 1) {
    if (tid < off) { s1[tid] += s1[tid + off]; s2[tid] += s2[tid + off]; }
    __syncthreads();
  }
  const float mu   = s1[0] * (1.0f / CCH);
  const float var  = s2[0] * (1.0f / CCH) - mu * mu;
  const float rstd = rsqrtf(var + 1e-5f);
  bf16_t* orow = out + (size_t)row * CCH;
  for (int c = tid; c < CCH; c += 256)
    orow[c] = (bf16_t)((xr[c] - mu) * rstd * g[c] + b[c]);
}

// ---------------------------------------------------------------------------
// WMMA GEMM with double-buffered async-LDS pipeline.
// Out[M,N] = epilogue( A[M,K](bf16) @ W[N,K]^T(bf16) )
// 256 threads = 8 waves; block tile 128x128; wave tile 64x32 (4x2 wmma tiles).
// Per 32-deep K slab: stage A(128x32)+B(128x32) once into LDS via
// global_load_async_to_lds_b128, prefetch next slab during compute.
// ---------------------------------------------------------------------------
template <int OUTBF, int GELU, int RESID, int CAUSAL>
__global__ __launch_bounds__(256) void gemm_wmma(
    const bf16_t* __restrict__ A, int lda, long long bsA,
    const bf16_t* __restrict__ W, int ldw, long long bsW,
    void* __restrict__ Out, int ldo, long long bsO,
    const float* __restrict__ Res, long long bsR,
    int Kdim) {
  __shared__ alignas(32) bf16_t Asm[2][128 * 32];
  __shared__ alignas(32) bf16_t Bsm[2][128 * 32];

  const int tid  = threadIdx.x;
  const int lane = tid & 31;
  const int wave = tid >> 5;              // 0..7
  const int wm   = wave >> 2;             // 0..1
  const int wn   = wave & 3;              // 0..3
  const int bz   = blockIdx.z;

  const bf16_t* Ab = A + (size_t)bsA * bz;
  const bf16_t* Wb = W + (size_t)bsW * bz;

  const int rowBlk = blockIdx.y * 128;
  const int colBlk = blockIdx.x * 128;

  int kend = Kdim;
  if (CAUSAL) {
    int kc = ((int)blockIdx.y + 1) * 128;
    if (kc < kend) kend = kc;
  }
  const int nk = kend >> 5;

  v8f acc[4][2];
  const v8f vz = {};
#pragma unroll
  for (int i = 0; i < 4; ++i) { acc[i][0] = vz; acc[i][1] = vz; }

  stage_tile_async(Ab, lda, rowBlk, 0, &Asm[0][0], tid);
  stage_tile_async(Wb, ldw, colBlk, 0, &Bsm[0][0], tid);

  int buf = 0;
  for (int ks = 0; ks < nk; ++ks) {
    wait_async_all();
    __syncthreads();                       // slab `buf` visible to all waves
    if (ks + 1 < nk) {                     // prefetch next slab during compute
      stage_tile_async(Ab, lda, rowBlk, (ks + 1) * 32, &Asm[buf ^ 1][0], tid);
      stage_tile_async(Wb, ldw, colBlk, (ks + 1) * 32, &Bsm[buf ^ 1][0], tid);
    }
    v16bf b0 = lds_b_frag(&Bsm[buf][0], wn * 32,      lane);
    v16bf b1 = lds_b_frag(&Bsm[buf][0], wn * 32 + 16, lane);
#pragma unroll
    for (int i = 0; i < 4; ++i) {
      v16bf a = lds_a_frag(&Asm[buf][0], wm * 64 + 16 * i, lane);
      acc[i][0] = wmma_bf16(a, b0, acc[i][0]);
      acc[i][1] = wmma_bf16(a, b1, acc[i][1]);
    }
    __syncthreads();                       // all waves done reading `buf`
    buf ^= 1;
  }

  // C/D layout: row = 16*i + v + 8*(lane>=16), col = 16*j + (lane&15)
  const int cl = lane & 15;
  const int rh = (lane >> 4) * 8;
#pragma unroll
  for (int i = 0; i < 4; ++i)
#pragma unroll
    for (int j = 0; j < 2; ++j)
#pragma unroll
      for (int v = 0; v < 8; ++v) {
        const int r = rowBlk + wm * 64 + 16 * i + rh + v;
        const int c = colBlk + wn * 32 + 16 * j + cl;
        float val = acc[i][j][v];
        if constexpr (GELU)
          val = 0.5f * val * (1.0f + erff(val * 0.70710678118654752f));
        if constexpr (RESID)
          val += Res[(size_t)bsR * bz + (size_t)r * ldo + c];
        if constexpr (OUTBF)
          ((bf16_t*)Out)[(size_t)bsO * bz + (size_t)r * ldo + c] = (bf16_t)val;
        else
          ((float*)Out)[(size_t)bsO * bz + (size_t)r * ldo + c] = val;
      }
}

// ---------------------------------------------------------------------------
// Attention pass 1: online softmax stats (row max m, row sumexp l).
// One wave per 16-row query tile; Q fragments re-read from global per key tile, K streamed.
// qkv layout: [B*T, 3C] bf16 ; Q at col 0, K at col C.
// ---------------------------------------------------------------------------
__global__ __launch_bounds__(32) void attn_stats(const bf16_t* __restrict__ qkv,
                                                 float* __restrict__ mbuf,
                                                 float* __restrict__ lbuf) {
  const int lane  = threadIdx.x;
  const int tTile = blockIdx.x;
  const int b     = blockIdx.y;
  const int t0    = tTile * 16;
  const bf16_t* Q = qkv + (size_t)b * TT * C3H;
  const bf16_t* K = Q + CCH;

  float m[8], l[8];
#pragma unroll
  for (int v = 0; v < 8; ++v) { m[v] = -3.0e38f; l[v] = 0.f; }

  const int cl = lane & 15;
  const int rh = (lane >> 4) * 8;

  for (int sT = 0; sT <= tTile; ++sT) {
    const int s0 = sT * 16;
    v8f acc = {};
    for (int k = 0; k < CCH; k += 32) {
      v16bf a  = load_a_frag(Q, C3H, t0, k, lane);
      v16bf bb = load_b_frag(K, C3H, s0, k, lane);
      acc = wmma_bf16(a, bb, acc);
    }
#pragma unroll
    for (int v = 0; v < 8; ++v) {
      const int r = rh + v;
      float s = acc[v] * ATT_SCALE;
      if (s0 + cl > t0 + r) s = -3.0e38f;          // causal (no-op off-diagonal)
      const float mx   = redmax16(s);
      const float newm = fmaxf(m[v], mx);
      const float se   = redsum16(__expf(s - newm));
      l[v] = l[v] * __expf(m[v] - newm) + se;
      m[v] = newm;
    }
  }
  if (cl == 0) {
#pragma unroll
    for (int v = 0; v < 8; ++v) {
      mbuf[(size_t)b * TT + t0 + rh + v] = m[v];
      lbuf[(size_t)b * TT + t0 + rh + v] = l[v];
    }
  }
}

// ---------------------------------------------------------------------------
// Attention pass 2: recompute score tiles, write normalized P (bf16).
// Zero-fills up to the 128-col block containing the diagonal so the causal
// P@V GEMM (K limit = (blockRow+1)*128) reads only initialized data.
// ---------------------------------------------------------------------------
__global__ __launch_bounds__(32) void attn_probs(const bf16_t* __restrict__ qkv,
                                                 const float* __restrict__ mbuf,
                                                 const float* __restrict__ lbuf,
                                                 bf16_t* __restrict__ P) {
  const int lane  = threadIdx.x;
  const int tTile = blockIdx.x;
  const int b     = blockIdx.y;
  const int t0    = tTile * 16;
  const bf16_t* Q = qkv + (size_t)b * TT * C3H;
  const bf16_t* K = Q + CCH;
  bf16_t* Pb = P + (size_t)b * TT * TT;

  const int cl = lane & 15;
  const int rh = (lane >> 4) * 8;

  float m[8], linv[8];
#pragma unroll
  for (int v = 0; v < 8; ++v) {
    m[v]    = mbuf[(size_t)b * TT + t0 + rh + v];
    linv[v] = 1.0f / lbuf[(size_t)b * TT + t0 + rh + v];
  }

  const int sEnd = ((tTile >> 3) + 1) << 3;    // round up to 128-col block
  for (int sT = 0; sT < sEnd; ++sT) {
    const int s0 = sT * 16;
    if (sT <= tTile) {
      v8f acc = {};
      for (int k = 0; k < CCH; k += 32) {
        v16bf a  = load_a_frag(Q, C3H, t0, k, lane);
        v16bf bb = load_b_frag(K, C3H, s0, k, lane);
        acc = wmma_bf16(a, bb, acc);
      }
#pragma unroll
      for (int v = 0; v < 8; ++v) {
        const int r = rh + v;
        float s = acc[v] * ATT_SCALE;
        if (s0 + cl > t0 + r) s = -3.0e38f;
        const float p = __expf(s - m[v]) * linv[v];
        Pb[(size_t)(t0 + r) * TT + s0 + cl] = (bf16_t)p;
      }
    } else {
#pragma unroll
      for (int v = 0; v < 8; ++v)
        Pb[(size_t)(t0 + rh + v) * TT + s0 + cl] = (bf16_t)0.0f;
    }
  }
}

// ---------------------------------------------------------------------------
// Transpose V[b][t][c] (inside qkv, col offset 2C) -> Vt[b][c][t], bf16, LDS tiled
// ---------------------------------------------------------------------------
__global__ __launch_bounds__(256) void transpose_v(const bf16_t* __restrict__ qkv,
                                                   bf16_t* __restrict__ vt) {
  __shared__ unsigned short tile[32][33];
  const int b  = blockIdx.z;
  const int t0 = blockIdx.x * 32;
  const int c0 = blockIdx.y * 32;
  const bf16_t* V = qkv + (size_t)b * TT * C3H + 2 * CCH;
#pragma unroll
  for (int j = 0; j < 4; ++j) {
    const int t = t0 + threadIdx.y + 8 * j;
    tile[threadIdx.y + 8 * j][threadIdx.x] =
        ((const unsigned short*)(V + (size_t)t * C3H))[c0 + threadIdx.x];
  }
  __syncthreads();
  bf16_t* Vt = vt + (size_t)b * CCH * TT;
#pragma unroll
  for (int j = 0; j < 4; ++j) {
    const int c = c0 + threadIdx.y + 8 * j;
    ((unsigned short*)(Vt + (size_t)c * TT))[t0 + threadIdx.x] =
        tile[threadIdx.x][threadIdx.y + 8 * j];
  }
}

// ---------------------------------------------------------------------------
// Host-side orchestration
// ---------------------------------------------------------------------------
extern "C" void kernel_launch(void* const* d_in, const int* in_sizes, int n_in,
                              void* d_out, int out_size, void* d_ws, size_t ws_size,
                              hipStream_t stream) {
  const float* x      = (const float*)d_in[0];
  const float* ln1_g  = (const float*)d_in[1];
  const float* ln1_b  = (const float*)d_in[2];
  const float* qkv_w  = (const float*)d_in[3];
  const float* proj_w = (const float*)d_in[4];
  const float* ln2_g  = (const float*)d_in[5];
  const float* ln2_b  = (const float*)d_in[6];
  const float* w1     = (const float*)d_in[7];
  const float* w2     = (const float*)d_in[8];
  float* out = (float*)d_out;

  // -------- workspace carve (256B aligned) --------
  size_t off = 0;
  auto carve = [&](size_t bytes) -> char* {
    char* p = (char*)d_ws + off;
    off += (bytes + 255) & ~(size_t)255;
    return p;
  };
  bf16_t* wqkv  = (bf16_t*)carve((size_t)C3H * CCH * 2);
  bf16_t* wproj = (bf16_t*)carve((size_t)CCH * CCH * 2);
  bf16_t* w1b   = (bf16_t*)carve((size_t)DFFH * CCH * 2);
  bf16_t* w2b   = (bf16_t*)carve((size_t)CCH * DFFH * 2);
  bf16_t* hbuf  = (bf16_t*)carve((size_t)BTH * CCH * 2);   // ln1 h / attn out / ln2 h
  float*  mbuf  = (float*)carve((size_t)BTH * 4);
  float*  lbuf  = (float*)carve((size_t)BTH * 4);
  bf16_t* qkvb  = (bf16_t*)carve((size_t)BTH * C3H * 2);
  const size_t pBytes  = (size_t)BB * TT * TT * 2;          // 134 MB
  const size_t vtBytes = (size_t)BB * CCH * TT * 2;         // 25 MB
  const size_t ffBytes = (size_t)BTH * DFFH * 2;            // 100 MB
  size_t r2 = pBytes + vtBytes; if (ffBytes > r2) r2 = ffBytes;
  char* region2 = carve(r2);
  bf16_t* Pbuf  = (bf16_t*)region2;
  bf16_t* Vtbuf = (bf16_t*)(region2 + pBytes);
  bf16_t* ffbuf = (bf16_t*)region2;   // aliases Pbuf: FFN runs after attention

  // -------- weights f32 -> bf16 --------
  {
    int n;
    n = C3H * CCH;  cvt_f32_bf16<<<(n + 255) / 256, 256, 0, stream>>>(qkv_w, wqkv, n);
    n = CCH * CCH;  cvt_f32_bf16<<<(n + 255) / 256, 256, 0, stream>>>(proj_w, wproj, n);
    n = DFFH * CCH; cvt_f32_bf16<<<(n + 255) / 256, 256, 0, stream>>>(w1, w1b, n);
    n = CCH * DFFH; cvt_f32_bf16<<<(n + 255) / 256, 256, 0, stream>>>(w2, w2b, n);
  }

  // -------- LN1 --------
  layernorm_bf16<<<BTH, 256, 0, stream>>>(x, ln1_g, ln1_b, hbuf);

  // -------- QKV GEMM: [BT,768] @ [2304,768]^T -> [BT,2304] bf16 --------
  gemm_wmma<1, 0, 0, 0><<<dim3(C3H / 128, BTH / 128, 1), 256, 0, stream>>>(
      hbuf, CCH, 0, wqkv, CCH, 0, qkvb, C3H, 0, nullptr, 0, CCH);

  // -------- attention softmax stats + probabilities --------
  attn_stats<<<dim3(TT / 16, BB), 32, 0, stream>>>(qkvb, mbuf, lbuf);
  attn_probs<<<dim3(TT / 16, BB), 32, 0, stream>>>(qkvb, mbuf, lbuf, Pbuf);

  // -------- V transpose, then attn = P @ Vt^T (causal K limit), bf16 out --------
  transpose_v<<<dim3(TT / 32, CCH / 32, BB), dim3(32, 8), 0, stream>>>(qkvb, Vtbuf);
  gemm_wmma<1, 0, 0, 1><<<dim3(CCH / 128, TT / 128, BB), 256, 0, stream>>>(
      Pbuf, TT, (long long)TT * TT, Vtbuf, TT, (long long)CCH * TT,
      hbuf, CCH, (long long)TT * CCH, nullptr, 0, TT);

  // -------- proj GEMM + residual(x) -> d_out (f32) --------
  gemm_wmma<0, 0, 1, 0><<<dim3(CCH / 128, BTH / 128, 1), 256, 0, stream>>>(
      hbuf, CCH, 0, wproj, CCH, 0, out, CCH, 0, x, 0, CCH);

  // -------- LN2 --------
  layernorm_bf16<<<BTH, 256, 0, stream>>>(out, ln2_g, ln2_b, hbuf);

  // -------- FFN1 + GELU -> bf16 --------
  gemm_wmma<1, 1, 0, 0><<<dim3(DFFH / 128, BTH / 128, 1), 256, 0, stream>>>(
      hbuf, CCH, 0, w1b, CCH, 0, ffbuf, DFFH, 0, nullptr, 0, CCH);

  // -------- FFN2 + residual(d_out) -> d_out (f32) --------
  gemm_wmma<0, 0, 1, 0><<<dim3(CCH / 128, BTH / 128, 1), 256, 0, stream>>>(
      ffbuf, DFFH, 0, w2b, DFFH, 0, out, CCH, 0, (const float*)out, 0, DFFH);

  (void)in_sizes; (void)n_in; (void)out_size; (void)ws_size;
}
